// ChamferLossKL_7155415515634
// MI455X (gfx1250) — matrix operand order repaired and checked by reading, compiled, not run on hardware
//
#include <hip/hip_runtime.h>
#include <hip/hip_bf16.h>
#include <stdint.h>

// ---------------------------------------------------------------------------
// Fused pairwise-KL chamfer loss for MI455X (gfx1250, wave32, WMMA + TDM).
// inner[x,y] = D + sum_la[x] - sum_lb[y] - t_mub[y] + A'[x,:]·B'[y,:]
// p_kl = -0.5*inner ; loss1 = sum_y min_x ; loss2 = sum_x mask*min_y
// min(p_kl) == -0.5 * max(inner)  -> track maxima with ordered-uint atomics.
// Packed operands are bf16 hi/lo split (K = 2*3*128 = 768) for ~f32 accuracy
// on the v_wmma_f32_16x16x32_bf16 pipe. Packed A+B ~100MB -> L2 resident.
// Tiles are fed by the Tensor Data Mover (tensor_load_to_lds) with hardware
// LDS padding reproducing the bank-conflict-free pitches (72 / 136 halves),
// double-buffered and synchronized on TENSORcnt.
// ---------------------------------------------------------------------------

typedef __attribute__((ext_vector_type(16))) __bf16 v16bf;
typedef __attribute__((ext_vector_type(8)))  float  v8f;
typedef __attribute__((ext_vector_type(4)))  unsigned int u32x4;
typedef __attribute__((ext_vector_type(8)))  int i32x8;
typedef __attribute__((ext_vector_type(4)))  int i32x4;

#define BS   8
#define NX   4096
#define NY   4096
#define DIM  128
#define KTOT 768           // 384 hi + 384 lo
#define BM   128
#define BN   128
#define KC   64
#define NCHUNK (KTOT / KC) // 12
#define PA   72            // LDS pitch (halves) for A tile: 32 dw rows + 4 dw pad
#define PB   136           // LDS pitch (halves) for B tile: 64 dw rows + 4 dw pad
#define ABUF (BM * PA)     // halves per A buffer
#define BBUFH (KC * PB)    // halves per B buffer
#define SMEM_BYTES (2 * ABUF * 2 + 2 * BBUFH * 2 + BN * 4)  // 72192 B
#define YT_PER_WG 8
#define NSTRIP 4

#if defined(__has_builtin)
#if __has_builtin(__builtin_amdgcn_tensor_load_to_lds) && \
    __has_builtin(__builtin_amdgcn_s_wait_tensorcnt)
#define USE_TDM 1
#endif
#endif
#ifndef USE_TDM
#define USE_TDM 0
#endif

__device__ __forceinline__ unsigned fkey(float f) {
  int b = __float_as_int(f);
  return (unsigned)(b ^ ((b >> 31) | 0x80000000));
}
__device__ __forceinline__ float fdec(unsigned u) {
  int b = (u & 0x80000000u) ? (int)(u ^ 0x80000000u) : ~(int)u;
  return __int_as_float(b);
}
__device__ __forceinline__ void bsplit(float x, unsigned short* h, unsigned short* l) {
  union { __bf16 b; unsigned short u; } c;
  __bf16 hb = (__bf16)x;
  c.b = hb; *h = c.u;
  c.b = (__bf16)(x - (float)hb); *l = c.u;
}

#if USE_TDM
// 2-D TDM tile load, D# per CDNA5 ISA 8.3/8.4 (count=1, type=2, data_size=2B,
// pad_enable with pad_interval/pad_amount producing the padded LDS pitch).
// 6-arg builtin form: (g0 u32x4, g1 i32x8, g2 i32x4, g3 i32x4, i32x8, cpol).
__device__ __forceinline__ void tdm_load_2d(unsigned lds_off, unsigned long long gaddr,
                                            unsigned g1d0, unsigned td0, unsigned td1,
                                            unsigned tile0, unsigned tile1,
                                            unsigned stride0) {
  u32x4 g0;
  g0[0] = 1u;                                              // count=1, user mode
  g0[1] = lds_off;                                         // LDS byte address
  g0[2] = (unsigned)(gaddr & 0xffffffffull);               // global addr lo
  g0[3] = (unsigned)((gaddr >> 32) & 0x01fffffful) | (2u << 30);  // hi | type=2
  i32x8 g1;
  g1[0] = (int)g1d0;                            // data_size|pad_en|pad codes
  g1[1] = (int)(td0 << 16);                     // abar=0 | tensor_dim0 lo
  g1[2] = (int)((td0 >> 16) | (td1 << 16));     // tensor_dim0 hi | tensor_dim1 lo
  g1[3] = (int)((td1 >> 16) | (tile0 << 16));   // tensor_dim1 hi | tile_dim0
  g1[4] = (int)tile1;                           // tile_dim1 | tile_dim2=0
  g1[5] = (int)stride0;                         // tensor_dim0_stride lo
  g1[6] = 0;                                    // stride0 hi | stride1 lo
  g1[7] = 0;
  i32x4 z4 = {0, 0, 0, 0};                      // 2-D tensor: groups 2/3 unused
  i32x8 z8 = {0, 0, 0, 0, 0, 0, 0, 0};
  __builtin_amdgcn_tensor_load_to_lds(g0, g1, z4, z4, z8, 0);
}
// group1 dword0: data_size=1(2B) | pad_enable | pad_interval code | pad_amount code
#define G1D0_A ((1u << 16) | (1u << 20) | (4u << 22) | (3u << 25))  // 32dw + 4dw pad
#define G1D0_B ((1u << 16) | (1u << 20) | (5u << 22) | (3u << 25))  // 64dw + 4dw pad
#endif

// --------------------------- pack A (preds side) ---------------------------
// A' rows: [mu | mu^2 | exp(la)] hi, then lo residuals. Layout [p][k], k fast.
__global__ __launch_bounds__(256)
void pack_a(const float* __restrict__ mu, const float* __restrict__ lv,
            unsigned short* __restrict__ Ag, float* __restrict__ rowoff,
            unsigned* __restrict__ rowmax) {
  const int wid = threadIdx.x >> 5, lane = threadIdx.x & 31;
  const int p = blockIdx.x * 8 + wid;          // (b,x) flattened
  const size_t base = (size_t)p * KTOT;
  float sla = 0.f;
  #pragma unroll
  for (int c = 0; c < 4; ++c) {
    const int d = lane + c * 32;
    const float m  = mu[(size_t)p * DIM + d];
    const float la = lv[(size_t)p * DIM + d];
    const float ea = __expf(la);
    sla += la;
    unsigned short h, l;
    bsplit(m, &h, &l);     Ag[base + d]       = h; Ag[base + 384 + d]       = l;
    bsplit(m * m, &h, &l); Ag[base + 128 + d] = h; Ag[base + 384 + 128 + d] = l;
    bsplit(ea, &h, &l);    Ag[base + 256 + d] = h; Ag[base + 384 + 256 + d] = l;
  }
  #pragma unroll
  for (int off = 16; off; off >>= 1) sla += __shfl_xor(sla, off, 32);
  if (lane == 0) { rowoff[p] = sla; rowmax[p] = 0u; }
}

// --------------------------- pack B (gts side) -----------------------------
// B' stored K-major: Bg[b][k][y] so the GEMM's LDS tile needs no transpose.
// k<128: 2*mu*inv ; 128..255 and 256..383: -inv ; +384: lo residuals.
__global__ __launch_bounds__(256)
void pack_b(const float* __restrict__ mu, const float* __restrict__ lv,
            unsigned short* __restrict__ Bg, float* __restrict__ coloff,
            unsigned* __restrict__ colmax) {
  __shared__ unsigned short h_s[DIM * 65];
  __shared__ unsigned short l_s[DIM * 65];
  __shared__ float slb_s[64], smub_s[64];
  const int b  = blockIdx.x >> 6;              // 64 blocks of 64 y per batch
  const int y0 = (blockIdx.x & 63) * 64;
  const int t  = threadIdx.x;
  if (t < 64) { slb_s[t] = 0.f; smub_s[t] = 0.f; }
  __syncthreads();
  const size_t inb  = ((size_t)b * NY + y0) * DIM;
  const size_t outb = (size_t)b * KTOT * NY;
  // pass 1: t1 = 2*mu*inv (+ per-y sums), transpose through LDS
  for (int it = 0; it < 32; ++it) {
    const int idx = t + it * 256;              // 64y x 128d
    const int yl = idx >> 7, d = idx & 127;    // d fast -> coalesced loads
    const float m  = mu[inb + (size_t)yl * DIM + d];
    const float lb = lv[inb + (size_t)yl * DIM + d];
    const float inv = __expf(-lb);
    atomicAdd(&slb_s[yl], lb);
    atomicAdd(&smub_s[yl], m * m * inv);
    unsigned short h, l;
    bsplit(2.f * m * inv, &h, &l);
    h_s[d * 65 + yl] = h; l_s[d * 65 + yl] = l;
  }
  __syncthreads();
  for (int it = 0; it < 32; ++it) {
    const int idx = t + it * 256;
    const int yl = idx & 63, d = idx >> 6;     // y fast -> coalesced stores
    Bg[outb + (size_t)d * NY + y0 + yl]         = h_s[d * 65 + yl];
    Bg[outb + (size_t)(384 + d) * NY + y0 + yl] = l_s[d * 65 + yl];
  }
  __syncthreads();
  // pass 2: t2 = -inv, duplicated at k=128+d and k=256+d
  for (int it = 0; it < 32; ++it) {
    const int idx = t + it * 256;
    const int yl = idx >> 7, d = idx & 127;
    const float inv = __expf(-lv[inb + (size_t)yl * DIM + d]);
    unsigned short h, l;
    bsplit(-inv, &h, &l);
    h_s[d * 65 + yl] = h; l_s[d * 65 + yl] = l;
  }
  __syncthreads();
  for (int it = 0; it < 32; ++it) {
    const int idx = t + it * 256;
    const int yl = idx & 63, d = idx >> 6;
    const unsigned short h = h_s[d * 65 + yl], l = l_s[d * 65 + yl];
    Bg[outb + (size_t)(128 + d) * NY + y0 + yl] = h;
    Bg[outb + (size_t)(256 + d) * NY + y0 + yl] = h;
    Bg[outb + (size_t)(512 + d) * NY + y0 + yl] = l;
    Bg[outb + (size_t)(640 + d) * NY + y0 + yl] = l;
  }
  if (t < 64) {
    coloff[(size_t)b * NY + y0 + t] = (float)DIM - slb_s[t] - smub_s[t];
    colmax[(size_t)b * NY + y0 + t] = 0u;      // fkey domain: 0 < any finite key
  }
}

// --------------------------- fused GEMM + max ------------------------------
__global__ __launch_bounds__(256)
void chamfer(const unsigned short* __restrict__ Ag,
             const unsigned short* __restrict__ Bg,
             const float* __restrict__ rowoff,
             const float* __restrict__ coloff,
             unsigned* __restrict__ rowmax,
             unsigned* __restrict__ colmax) {
  extern __shared__ unsigned char smem_raw[];
  unsigned short* ldsA = (unsigned short*)smem_raw;                     // [2][ABUF]
  unsigned short* ldsB = (unsigned short*)(smem_raw + 2 * ABUF * 2);    // [2][BBUFH]
  unsigned* cmax_s = (unsigned*)(smem_raw + 2 * ABUF * 2 + 2 * BBUFH * 2);

  const int b     = blockIdx.z;
  const int x0    = blockIdx.x * BM;
  const int ybase = blockIdx.y * (YT_PER_WG * BN);
  const int t     = threadIdx.x;
  const int lane  = t & 31;
  const int wid   = t >> 5;
  const int waveM = wid & 3, waveN = wid >> 2; // 4x2 wave grid: 32x64 per wave
  const int hi    = lane >> 4, ln = lane & 15;

  // row offsets for this wave's 32 rows (C layout: M = 16i + 8*hi + v)
  float ro[2][8];
  #pragma unroll
  for (int i = 0; i < 2; ++i)
    #pragma unroll
    for (int v = 0; v < 8; ++v)
      ro[i][v] = rowoff[(size_t)b * NX + x0 + waveM * 32 + i * 16 + hi * 8 + v];

  float rm[2][8];
  #pragma unroll
  for (int i = 0; i < 2; ++i)
    #pragma unroll
    for (int v = 0; v < 8; ++v) rm[i][v] = -3.4e38f;

  const size_t abase = ((size_t)b * NX + x0) * KTOT;
  const size_t bbase = (size_t)b * KTOT * NY;
#if USE_TDM
  const unsigned ldsA_off = (unsigned)(size_t)ldsA;
  const unsigned ldsB_off = (unsigned)(size_t)ldsB;
#else
  const int arow = t >> 1, aseg = t & 1;       // A: 128 rows x 64 halves
  const int brow = t >> 2, bseg = t & 3;       // B: 64 k-rows x 128 halves
#endif

  for (int yt = 0; yt < YT_PER_WG; ++yt) {
    const int y0 = ybase + yt * BN;
    if (t < BN) cmax_s[t] = 0u;
    float co[4];
    #pragma unroll
    for (int j = 0; j < 4; ++j)
      co[j] = coloff[(size_t)b * NY + y0 + waveN * 64 + j * 16 + ln];

    v8f acc[2][4];
    #pragma unroll
    for (int i = 0; i < 2; ++i)
      #pragma unroll
      for (int j = 0; j < 4; ++j) {
        v8f z = {0.f, 0.f, 0.f, 0.f, 0.f, 0.f, 0.f, 0.f};
        acc[i][j] = z;
      }

#if USE_TDM
    if (wid == 0) {  // prologue: DMA chunk 0 into buffer 0 (TDM ignores EXEC)
      tdm_load_2d(ldsA_off, (unsigned long long)(size_t)(Ag + abase),
                  G1D0_A, KTOT, NX, KC, BM, KTOT);
      tdm_load_2d(ldsB_off, (unsigned long long)(size_t)(Bg + bbase + y0),
                  G1D0_B, NY, KTOT, BN, KC, NY);
    }
#endif
    for (int c = 0; c < NCHUNK; ++c) {
      const int cur = c & 1;
      const unsigned short* bufA = ldsA + cur * ABUF;
      const unsigned short* bufB = ldsB + cur * BBUFH;
#if USE_TDM
      if (wid == 0) {
        if (c + 1 < NCHUNK) {  // DMA next chunk into the other buffer
          const int nc = c + 1, nb = nc & 1;
          tdm_load_2d(ldsA_off + (unsigned)(nb * ABUF * 2),
                      (unsigned long long)(size_t)(Ag + abase + nc * KC),
                      G1D0_A, KTOT, NX, KC, BM, KTOT);
          tdm_load_2d(ldsB_off + (unsigned)(nb * BBUFH * 2),
                      (unsigned long long)(size_t)(Bg + bbase + (size_t)(nc * KC) * NY + y0),
                      G1D0_B, NY, KTOT, BN, KC, NY);
          __builtin_amdgcn_s_wait_tensorcnt(2);  // chunk c landed (in-order)
        } else {
          __builtin_amdgcn_s_wait_tensorcnt(0);
        }
      }
      __syncthreads();
#else
      __syncthreads();
      {
        const unsigned short* src = Ag + abase + (size_t)arow * KTOT + c * KC + aseg * 32;
        unsigned short* dst = (unsigned short*)bufA + arow * PA + aseg * 32;
        #pragma unroll
        for (int j = 0; j < 4; ++j)
          *(uint4*)(dst + j * 8) = *(const uint4*)(src + j * 8);
      }
      {
        const unsigned short* src = Bg + bbase + (size_t)(c * KC + brow) * NY + y0 + bseg * 32;
        unsigned short* dst = (unsigned short*)bufB + brow * PB + bseg * 32;
        #pragma unroll
        for (int j = 0; j < 4; ++j)
          *(uint4*)(dst + j * 8) = *(const uint4*)(src + j * 8);
      }
      __syncthreads();
#endif
      #pragma unroll
      for (int kk = 0; kk < KC; kk += 32) {
        // A frag (ISA 16-bit A 16x32 layout): elems 0-7 = K kk+8*hi+0..7,
        // elems 8-15 = K kk+16+8*hi+0..7; lane&15 selects row M.
        union { uint4 q[2]; v16bf v; } afr[2], bfr;
        #pragma unroll
        for (int i = 0; i < 2; ++i) {
          const unsigned short* pa = bufA + (waveM * 32 + i * 16 + ln) * PA + kk + hi * 8;
          afr[i].q[0] = *(const uint4*)pa;
          afr[i].q[1] = *(const uint4*)(pa + 16);
        }
        // B frag: lane&15 = N, elems 0-7 = K kk+8*hi+0..7, elems 8-15 = K kk+16+8*hi+0..7.
        #pragma unroll
        for (int j = 0; j < 4; ++j) {
          const unsigned short* pb = bufB + (kk + hi * 8) * PB + waveN * 64 + j * 16 + ln;
          #pragma unroll
          for (int s = 0; s < 8; ++s) {
            ((unsigned short*)&bfr)[s]     = pb[s * PB];
            ((unsigned short*)&bfr)[8 + s] = pb[(16 + s) * PB];
          }
          #pragma unroll
          for (int i = 0; i < 2; ++i)
            acc[i][j] = __builtin_amdgcn_wmma_f32_16x16x32_bf16(
                false, afr[i].v, false, bfr.v, (short)0, acc[i][j], false, false);
        }
      }
#if USE_TDM
      __syncthreads();  // all LDS reads of chunk c done before its buffer is reused
#endif
    }
    // epilogue: col-max needs +rowoff before max over rows; row-max needs
    // +coloff before max over cols. The other offset is added at flush time.
    #pragma unroll
    for (int j = 0; j < 4; ++j) {
      float cm = -3.4e38f;
      #pragma unroll
      for (int i = 0; i < 2; ++i)
        #pragma unroll
        for (int v = 0; v < 8; ++v) {
          const float val = acc[i][j][v];
          cm = fmaxf(cm, val + ro[i][v]);
          rm[i][v] = fmaxf(rm[i][v], val + co[j]);
        }
      cm = fmaxf(cm, __shfl_xor(cm, 16, 32));  // combine M halves (same N)
      if (hi == 0) atomicMax(&cmax_s[waveN * 64 + j * 16 + ln], fkey(cm));
    }
    __syncthreads();
    if (t < BN) {
      const float g = fdec(cmax_s[t]) + coloff[(size_t)b * NY + y0 + t];
      atomicMax(&colmax[(size_t)b * NY + y0 + t], fkey(g));
    }
  }
  // row-max flush: butterfly over the 16 lanes sharing the same M rows.
  #pragma unroll
  for (int i = 0; i < 2; ++i)
    #pragma unroll
    for (int v = 0; v < 8; ++v) {
      float r = rm[i][v];
      r = fmaxf(r, __shfl_xor(r, 1, 32));
      r = fmaxf(r, __shfl_xor(r, 2, 32));
      r = fmaxf(r, __shfl_xor(r, 4, 32));
      r = fmaxf(r, __shfl_xor(r, 8, 32));
      rm[i][v] = r;
    }
  if (ln == 0) {
    #pragma unroll
    for (int i = 0; i < 2; ++i)
      #pragma unroll
      for (int v = 0; v < 8; ++v) {
        const int m = x0 + waveM * 32 + i * 16 + hi * 8 + v;
        atomicMax(&rowmax[(size_t)b * NX + m], fkey(rm[i][v] + ro[i][v]));
      }
  }
}

// --------------------------- final reduction -------------------------------
__global__ __launch_bounds__(256)
void reduce_out(const unsigned* __restrict__ rowmax,
                const unsigned* __restrict__ colmax,
                const float* __restrict__ mask, float* __restrict__ out) {
  __shared__ float red[256];
  const int b = blockIdx.x, t = threadIdx.x;
  float s = 0.f;
  for (int y = t; y < NY; y += 256) s += -0.5f * fdec(colmax[(size_t)b * NY + y]);
  for (int x = t; x < NX; x += 256)
    s += -0.5f * fdec(rowmax[(size_t)b * NX + x]) * mask[(size_t)b * NX + x];
  red[t] = s;
  __syncthreads();
  for (int o = 128; o; o >>= 1) {
    if (t < o) red[t] += red[t + o];
    __syncthreads();
  }
  if (t == 0) out[b] = red[0];
}

extern "C" void kernel_launch(void* const* d_in, const int* in_sizes, int n_in,
                              void* d_out, int out_size, void* d_ws, size_t ws_size,
                              hipStream_t stream) {
  (void)in_sizes; (void)n_in; (void)out_size; (void)ws_size;
  const float* mu_a = (const float*)d_in[0];
  const float* lv_a = (const float*)d_in[1];
  const float* mu_b = (const float*)d_in[2];
  const float* lv_b = (const float*)d_in[3];
  const float* mask = (const float*)d_in[4];

  char* ws = (char*)d_ws;
  const size_t szA  = (size_t)BS * NX * KTOT * sizeof(unsigned short); // 50.3 MB
  const size_t szB  = (size_t)BS * KTOT * NY * sizeof(unsigned short); // 50.3 MB
  const size_t szV  = (size_t)BS * NX * sizeof(float);                 // 128 KB
  unsigned short* Ag = (unsigned short*)(ws);
  unsigned short* Bg = (unsigned short*)(ws + szA);
  float* rowoff  = (float*)(ws + szA + szB);
  float* coloff  = (float*)(ws + szA + szB + szV);
  unsigned* rmax = (unsigned*)(ws + szA + szB + 2 * szV);
  unsigned* cmax = (unsigned*)(ws + szA + szB + 3 * szV);

  pack_a<<<dim3((BS * NX) / 8), dim3(256), 0, stream>>>(mu_a, lv_a, Ag, rowoff, rmax);
  pack_b<<<dim3(BS * (NY / 64)), dim3(256), 0, stream>>>(mu_b, lv_b, Bg, coloff, cmax);
  chamfer<<<dim3(NX / BM, NSTRIP, BS), dim3(256), SMEM_BYTES, stream>>>(
      Ag, Bg, rowoff, coloff, rmax, cmax);
  reduce_out<<<dim3(BS), dim3(256), 0, stream>>>(rmax, cmax, mask, (float*)d_out);
}
